// Net_26963804684991
// MI455X (gfx1250) — matrix operand order repaired and checked by reading, not compiled, not run on hardware
//
#include <hip/hip_runtime.h>
#include <hip/hip_bf16.h>
#include <hip/hip_fp16.h>

typedef __attribute__((ext_vector_type(16))) _Float16 v16h;
typedef __attribute__((ext_vector_type(8)))  float    v8f;

#define B_   32
#define NPG_ 2048
#define NN_  (B_*NPG_)        // 65536
#define DEG_ 8
#define EE_  (NN_*DEG_)       // 524288
#define INF_ 64
#define H1_  128
#define H2_  256
#define K1_  (NPG_/2)         // 1024
#define K2_  (K1_/2)          // 512
#define EPS_ 1e-5f

__device__ __forceinline__ void atomAddF(float* p, float v) {
  unsafeAtomicAdd(p, v);   // lowers to global_atomic_add_f32
}

// ---------- weight prep: f32 [Fin,Fout] -> f16 transposed [Fout,Fin] ----------
__global__ void wtrans_kernel(const float* __restrict__ W, _Float16* __restrict__ WT,
                              int Fin, int Fout) {
  int tid = blockIdx.x * blockDim.x + threadIdx.x;
  if (tid >= Fin * Fout) return;
  int k = tid / Fout, n = tid % Fout;
  WT[(size_t)n * Fin + k] = (_Float16)W[(size_t)k * Fout + n];
}

// ---------- edge scatter: acc[tgt] += x[src]*m (feature-parallel) ----------
template<int F>
__global__ void scatter_feat_kernel(const float* __restrict__ x,
                                    const int* __restrict__ src, const int* __restrict__ tgt,
                                    const float* __restrict__ em,
                                    float* __restrict__ acc, int E) {
  size_t tid = (size_t)blockIdx.x * blockDim.x + threadIdx.x;
  if (tid >= (size_t)E * F) return;
  int e = (int)(tid / F), f = (int)(tid % F);      // F is power of two -> shift/mask
  float m = em ? em[e] : 1.0f;
  if (m != 0.0f)
    atomAddF(&acc[(size_t)tgt[e] * F + f], x[(size_t)src[e] * F + f] * m);
}

__global__ void scatter_cnt_kernel(const int* __restrict__ tgt, const float* __restrict__ em,
                                   float* __restrict__ cnt, int E) {
  int e = blockIdx.x * blockDim.x + threadIdx.x;
  if (e >= E) return;
  float m = em ? em[e] : 1.0f;
  if (m != 0.0f) atomAddF(&cnt[tgt[e]], m);
}

// ---------- fused SAGEConv GEMM: Out = (acc/max(cnt,1))@Wl + Xr@Wr + bias ----------
// One wave per 16-row M-slab. All A fragments (both matrices, all K steps) are
// kept in VGPRs; the wave loops over N tiles reloading only the small
// L2-resident B fragments. v_wmma_f32_16x16x32_f16, fp32 accumulate.
template<int FIN>
__global__ __launch_bounds__(32)
void sage_gemm_wmma(const float* __restrict__ Xa, const float* __restrict__ cnt,
                    const float* __restrict__ Xr,
                    const _Float16* __restrict__ WlT, const _Float16* __restrict__ WrT,
                    const float* __restrict__ bias,
                    float* __restrict__ Out, int Fout) {
  constexpr int KS = FIN / 32;
  int tm = blockIdx.x;
  int lane = threadIdx.x;
  int l15 = lane & 15, half = lane >> 4;
  int row = tm * 16 + l15;            // A row this lane supplies
  float inv = 1.0f / fmaxf(cnt[row], 1.0f);

  // Load + convert A fragments once (16x32 f16 layout: elems i -> K=k0+half*16+i)
  v16h aA[KS], aR[KS];
  const float* xa_row = Xa + (size_t)row * FIN;
  const float* xr_row = Xr + (size_t)row * FIN;
#pragma unroll
  for (int ks = 0; ks < KS; ks++) {
    const float* p0 = xa_row + ks * 32 + half * 16;
    const float* p1 = p0 + 8;
    const float* q0 = xr_row + ks * 32 + half * 16;
    const float* q1 = q0 + 8;
#pragma unroll
    for (int i = 0; i < 8; i++) {
      aA[ks][i]     = (_Float16)(p0[i] * inv);
      aA[ks][8 + i] = (_Float16)(p1[i] * inv);
      aR[ks][i]     = (_Float16)q0[i];
      aR[ks][8 + i] = (_Float16)q1[i];
    }
  }

  float* out_base = Out + (size_t)(tm * 16 + half * 8) * Fout + l15;
  for (int tn = 0; tn < Fout / 16; tn++) {
    int col = tn * 16 + l15;          // C/D column this lane holds
    v8f acc;
    float bv = bias[col];
#pragma unroll
    for (int i = 0; i < 8; i++) acc[i] = bv;

    const _Float16* wl = WlT + (size_t)col * FIN + half * 16;
    const _Float16* wr = WrT + (size_t)col * FIN + half * 16;
#pragma unroll
    for (int ks = 0; ks < KS; ks++) {
      v16h b = *(const v16h*)(wl + ks * 32);   // B 32x16: elems i -> K = k0 + half*16 + i
      acc = __builtin_amdgcn_wmma_f32_16x16x32_f16(false, aA[ks], false, b, (short)0, acc, false, false);
      b = *(const v16h*)(wr + ks * 32);
      acc = __builtin_amdgcn_wmma_f32_16x16x32_f16(false, aR[ks], false, b, (short)0, acc, false, false);
    }
    // D 16x16 f32: VGPR r -> row = half*8 + r
#pragma unroll
    for (int r = 0; r < 8; r++)
      out_base[(size_t)r * Fout + tn * 16] = acc[r];
  }
}

// ---------- batchnorm ----------
#define BN_SLABS 64
template<int F>
__global__ void bn_reduce_kernel(const float* __restrict__ h, float* __restrict__ sums,
                                 float* __restrict__ sumsq, int M) {
  int tid = blockIdx.x * blockDim.x + threadIdx.x;
  if (tid >= F * BN_SLABS) return;
  int c = tid % F, slab = tid / F;
  int rows = M / BN_SLABS;
  const float* p = h + (size_t)slab * rows * F + c;
  float s = 0.f, q = 0.f;
  for (int r = 0; r < rows; r++) { float v = p[(size_t)r * F]; s += v; q += v * v; }
  atomAddF(&sums[c], s);
  atomAddF(&sumsq[c], q);
}

__global__ void bn_finalize_kernel(const float* __restrict__ sums, const float* __restrict__ sumsq,
                                   const float* __restrict__ g, const float* __restrict__ b,
                                   float* __restrict__ scale, float* __restrict__ shift,
                                   int M, int F) {
  int c = blockIdx.x * blockDim.x + threadIdx.x;
  if (c >= F) return;
  float m = sums[c] / (float)M;
  float v = sumsq[c] / (float)M - m * m;
  float sc = g[c] * rsqrtf(v + EPS_);
  scale[c] = sc;
  shift[c] = b[c] - m * sc;
}

template<int F>
__global__ void bn_apply_relu_kernel(float* __restrict__ h, const float* __restrict__ scale,
                                     const float* __restrict__ shift, int M) {
  size_t tid = (size_t)blockIdx.x * blockDim.x + threadIdx.x;
  if (tid >= (size_t)M * F) return;
  int c = (int)(tid % F);
  h[tid] = fmaxf(h[tid] * scale[c] + shift[c], 0.0f);
}

// ---------- score GEMV: t = h@Wrel ; s = h@Wroot + brel ----------
template<int F>
__global__ void score_gemv_kernel(const float* __restrict__ h, const float* __restrict__ Wrel,
                                  const float* __restrict__ Wroot, const float* __restrict__ brel,
                                  float* __restrict__ t, float* __restrict__ s, int M) {
  int n = blockIdx.x * blockDim.x + threadIdx.x;
  if (n >= M) return;
  const float* row = h + (size_t)n * F;
  float a = 0.f, r = 0.f;
#pragma unroll 8
  for (int k = 0; k < F; k++) { float v = row[k]; a += v * Wrel[k]; r += v * Wroot[k]; }
  t[n] = a;
  s[n] = r + brel[0];
}

__global__ void scatter_scalar_kernel(const float* __restrict__ t,
                                      const int* __restrict__ src, const int* __restrict__ tgt,
                                      const float* __restrict__ em, float* __restrict__ s, int E) {
  int e = blockIdx.x * blockDim.x + threadIdx.x;
  if (e >= E) return;
  float m = em ? em[e] : 1.0f;
  if (m != 0.0f) atomAddF(&s[tgt[e]], t[src[e]] * m);
}

// ---------- per-graph top-k via rank counting (matches lax.top_k ordering) ----------
__global__ void topk_rank_kernel(const float* __restrict__ score, int npg, int k,
                                 int* __restrict__ perm, float* __restrict__ vals) {
  int g = blockIdx.x;
  const float* s = score + (size_t)g * npg;
  __shared__ float tile[256];
  for (int i = threadIdx.x; i < npg; i += 256) {
    float si = s[i];
    int rank = 0;
    for (int j0 = 0; j0 < npg; j0 += 256) {
      tile[threadIdx.x] = s[j0 + threadIdx.x];
      __syncthreads();
      for (int jj = 0; jj < 256; jj++) {
        float sj = tile[jj];
        int j = j0 + jj;
        rank += (int)((sj > si) || (sj == si && j < i));
      }
      __syncthreads();
    }
    if (rank < k) { perm[(size_t)g * k + rank] = g * npg + i; vals[(size_t)g * k + rank] = si; }
  }
}

__global__ void build_map_kernel(const int* __restrict__ perm, int* __restrict__ map, int kn) {
  int j = blockIdx.x * blockDim.x + threadIdx.x;
  if (j >= kn) return;
  map[perm[j]] = j;
}

template<int F>
__global__ void gather_pool_kernel(const float* __restrict__ h, const int* __restrict__ perm,
                                   const float* __restrict__ vals, float* __restrict__ xo,
                                   int kn) {
  size_t tid = (size_t)blockIdx.x * blockDim.x + threadIdx.x;
  if (tid >= (size_t)kn * F) return;
  int j = (int)(tid / F), f = (int)(tid % F);
  xo[tid] = h[(size_t)perm[j] * F + f] * tanhf(vals[j]);
}

__global__ void edge_remap_kernel(const int* __restrict__ src, const int* __restrict__ tgt,
                                  const float* __restrict__ em, const int* __restrict__ map,
                                  int* __restrict__ so, int* __restrict__ to,
                                  float* __restrict__ eo, int E) {
  int e = blockIdx.x * blockDim.x + threadIdx.x;
  if (e >= E) return;
  int ns = map[src[e]], nt = map[tgt[e]];
  float m = em ? em[e] : 1.0f;
  eo[e] = (ns >= 0 && nt >= 0) ? m : 0.0f;
  so[e] = ns < 0 ? 0 : ns;
  to[e] = nt < 0 ? 0 : nt;
}

// ---------- final readout: mean-pool -> relu -> linear(256,2) -> softmax ----------
__global__ __launch_bounds__(256)
void readout_kernel(const float* __restrict__ h, const float* __restrict__ Wlin,
                    const float* __restrict__ blin, float* __restrict__ out, int k2, int F) {
  int g = blockIdx.x;
  int c = threadIdx.x;                 // F == 256 == blockDim
  const float* base = h + (size_t)g * k2 * F + c;
  float sum = 0.f;
  for (int i = 0; i < k2; i++) sum += base[(size_t)i * F];
  float p = fmaxf(sum / (float)k2, 0.0f);
  __shared__ float r0[256], r1[256];
  r0[c] = p * Wlin[c * 2 + 0];
  r1[c] = p * Wlin[c * 2 + 1];
  __syncthreads();
  for (int st = 128; st > 0; st >>= 1) {
    if (c < st) { r0[c] += r0[c + st]; r1[c] += r1[c + st]; }
    __syncthreads();
  }
  if (c == 0) {
    float o0 = r0[0] + blin[0], o1 = r1[0] + blin[1];
    float mx = fmaxf(o0, o1);
    float e0 = expf(o0 - mx), e1 = expf(o1 - mx);
    float d = e0 + e1;
    out[g * 2 + 0] = e0 / d;
    out[g * 2 + 1] = e1 / d;
  }
}

static inline unsigned cdivu(size_t a, size_t b) { return (unsigned)((a + b - 1) / b); }

extern "C" void kernel_launch(void* const* d_in, const int* in_sizes, int n_in,
                              void* d_out, int out_size, void* d_ws, size_t ws_size,
                              hipStream_t stream) {
  const float* x      = (const float*)d_in[0];
  const int*   ei     = (const int*)d_in[1];
  const int*   src    = ei;
  const int*   tgt    = ei + EE_;
  const float* W1l    = (const float*)d_in[3];
  const float* b1l    = (const float*)d_in[4];
  const float* W1r    = (const float*)d_in[5];
  const float* bn1g   = (const float*)d_in[6];
  const float* bn1b   = (const float*)d_in[7];
  const float* p1Wrel = (const float*)d_in[8];
  const float* p1brel = (const float*)d_in[9];
  const float* p1Wroot= (const float*)d_in[10];
  const float* W2l    = (const float*)d_in[11];
  const float* b2l    = (const float*)d_in[12];
  const float* W2r    = (const float*)d_in[13];
  const float* bn2g   = (const float*)d_in[14];
  const float* bn2b   = (const float*)d_in[15];
  const float* p2Wrel = (const float*)d_in[16];
  const float* p2brel = (const float*)d_in[17];
  const float* p2Wroot= (const float*)d_in[18];
  const float* Wlin   = (const float*)d_in[19];
  const float* blin   = (const float*)d_in[20];
  float* out = (float*)d_out;

  char* ws = (char*)d_ws;
  size_t off = 0;
  auto alloc = [&](size_t bytes) -> char* {
    char* p = ws + off;
    off = (off + bytes + 255) & ~(size_t)255;
    return p;
  };

  _Float16* wt1l = (_Float16*)alloc((size_t)INF_ * H1_ * 2);
  _Float16* wt1r = (_Float16*)alloc((size_t)INF_ * H1_ * 2);
  _Float16* wt2l = (_Float16*)alloc((size_t)H1_ * H2_ * 2);
  _Float16* wt2r = (_Float16*)alloc((size_t)H1_ * H2_ * 2);
  float* acc1 = (float*)alloc((size_t)NN_ * INF_ * 4);
  float* cnt1 = (float*)alloc((size_t)NN_ * 4);
  float* h1   = (float*)alloc((size_t)NN_ * H1_ * 4);
  float* bnS  = (float*)alloc(256 * 4);
  float* bnQ  = (float*)alloc(256 * 4);
  float* bnSc = (float*)alloc(256 * 4);
  float* bnSh = (float*)alloc(256 * 4);
  float* t1   = (float*)alloc((size_t)NN_ * 4);
  float* s1   = (float*)alloc((size_t)NN_ * 4);
  int*   perm1= (int*)alloc((size_t)B_ * K1_ * 4);
  float* vals1= (float*)alloc((size_t)B_ * K1_ * 4);
  int*   map1 = (int*)alloc((size_t)NN_ * 4);
  float* h1p  = (float*)alloc((size_t)B_ * K1_ * H1_ * 4);
  int*   src2 = (int*)alloc((size_t)EE_ * 4);
  int*   tgt2 = (int*)alloc((size_t)EE_ * 4);
  float* em2  = (float*)alloc((size_t)EE_ * 4);
  float* acc2 = (float*)alloc((size_t)B_ * K1_ * H2_ * 4);
  float* cnt2 = (float*)alloc((size_t)B_ * K1_ * 4);
  float* h2   = (float*)alloc((size_t)B_ * K1_ * H2_ * 4);
  float* t2   = (float*)alloc((size_t)B_ * K1_ * 4);
  float* s2   = (float*)alloc((size_t)B_ * K1_ * 4);
  int*   perm2= (int*)alloc((size_t)B_ * K2_ * 4);
  float* vals2= (float*)alloc((size_t)B_ * K2_ * 4);
  float* h2p  = (float*)alloc((size_t)B_ * K2_ * H2_ * 4);
  (void)ws_size; (void)in_sizes; (void)n_in; (void)out_size;

  const int TB = 256;

  // ---- weight prep (f16, transposed) ----
  wtrans_kernel<<<cdivu((size_t)INF_ * H1_, TB), TB, 0, stream>>>(W1l, wt1l, INF_, H1_);
  wtrans_kernel<<<cdivu((size_t)INF_ * H1_, TB), TB, 0, stream>>>(W1r, wt1r, INF_, H1_);
  wtrans_kernel<<<cdivu((size_t)H1_ * H2_, TB), TB, 0, stream>>>(W2l, wt2l, H1_, H2_);
  wtrans_kernel<<<cdivu((size_t)H1_ * H2_, TB), TB, 0, stream>>>(W2r, wt2r, H1_, H2_);

  // ---- layer 1: mean aggregation ----
  hipMemsetAsync(acc1, 0, (size_t)NN_ * INF_ * 4, stream);
  hipMemsetAsync(cnt1, 0, (size_t)NN_ * 4, stream);
  scatter_feat_kernel<INF_><<<cdivu((size_t)EE_ * INF_, TB), TB, 0, stream>>>(x, src, tgt, nullptr, acc1, EE_);
  scatter_cnt_kernel<<<cdivu(EE_, TB), TB, 0, stream>>>(tgt, nullptr, cnt1, EE_);

  // ---- layer 1: fused SAGE GEMM (WMMA) ----
  sage_gemm_wmma<INF_><<<NN_ / 16, 32, 0, stream>>>(acc1, cnt1, x, wt1l, wt1r, b1l, h1, H1_);

  // ---- layer 1: batchnorm + relu ----
  hipMemsetAsync(bnS, 0, 256 * 4, stream);
  hipMemsetAsync(bnQ, 0, 256 * 4, stream);
  bn_reduce_kernel<H1_><<<cdivu((size_t)H1_ * BN_SLABS, TB), TB, 0, stream>>>(h1, bnS, bnQ, NN_);
  bn_finalize_kernel<<<1, H1_, 0, stream>>>(bnS, bnQ, bn1g, bn1b, bnSc, bnSh, NN_, H1_);
  bn_apply_relu_kernel<H1_><<<cdivu((size_t)NN_ * H1_, TB), TB, 0, stream>>>(h1, bnSc, bnSh, NN_);

  // ---- layer 1: SAGPool score (reassociated GEMV + scalar scatter) ----
  score_gemv_kernel<H1_><<<cdivu(NN_, TB), TB, 0, stream>>>(h1, p1Wrel, p1Wroot, p1brel, t1, s1, NN_);
  scatter_scalar_kernel<<<cdivu(EE_, TB), TB, 0, stream>>>(t1, src, tgt, nullptr, s1, EE_);

  // ---- layer 1: top-k pool ----
  topk_rank_kernel<<<B_, TB, 0, stream>>>(s1, NPG_, K1_, perm1, vals1);
  hipMemsetAsync(map1, 0xFF, (size_t)NN_ * 4, stream);
  build_map_kernel<<<cdivu((size_t)B_ * K1_, TB), TB, 0, stream>>>(perm1, map1, B_ * K1_);
  gather_pool_kernel<H1_><<<cdivu((size_t)B_ * K1_ * H1_, TB), TB, 0, stream>>>(h1, perm1, vals1, h1p, B_ * K1_);
  edge_remap_kernel<<<cdivu(EE_, TB), TB, 0, stream>>>(src, tgt, nullptr, map1, src2, tgt2, em2, EE_);

  // ---- layer 2: masked mean aggregation ----
  hipMemsetAsync(acc2, 0, (size_t)B_ * K1_ * H1_ * 4, stream);
  hipMemsetAsync(cnt2, 0, (size_t)B_ * K1_ * 4, stream);
  scatter_feat_kernel<H1_><<<cdivu((size_t)EE_ * H1_, TB), TB, 0, stream>>>(h1p, src2, tgt2, em2, acc2, EE_);
  scatter_cnt_kernel<<<cdivu(EE_, TB), TB, 0, stream>>>(tgt2, em2, cnt2, EE_);

  // ---- layer 2: fused SAGE GEMM (WMMA) ----
  sage_gemm_wmma<H1_><<<(B_ * K1_) / 16, 32, 0, stream>>>(acc2, cnt2, h1p, wt2l, wt2r, b2l, h2, H2_);

  // ---- layer 2: batchnorm + relu ----
  hipMemsetAsync(bnS, 0, 256 * 4, stream);
  hipMemsetAsync(bnQ, 0, 256 * 4, stream);
  bn_reduce_kernel<H2_><<<cdivu((size_t)H2_ * BN_SLABS, TB), TB, 0, stream>>>(h2, bnS, bnQ, B_ * K1_);
  bn_finalize_kernel<<<1, H2_, 0, stream>>>(bnS, bnQ, bn2g, bn2b, bnSc, bnSh, B_ * K1_, H2_);
  bn_apply_relu_kernel<H2_><<<cdivu((size_t)B_ * K1_ * H2_, TB), TB, 0, stream>>>(h2, bnSc, bnSh, B_ * K1_);

  // ---- layer 2: score + top-k pool ----
  score_gemv_kernel<H2_><<<cdivu((size_t)B_ * K1_, TB), TB, 0, stream>>>(h2, p2Wrel, p2Wroot, p2brel, t2, s2, B_ * K1_);
  scatter_scalar_kernel<<<cdivu(EE_, TB), TB, 0, stream>>>(t2, src2, tgt2, em2, s2, EE_);
  topk_rank_kernel<<<B_, TB, 0, stream>>>(s2, K1_, K2_, perm2, vals2);
  gather_pool_kernel<H2_><<<cdivu((size_t)B_ * K2_ * H2_, TB), TB, 0, stream>>>(h2, perm2, vals2, h2p, B_ * K2_);

  // ---- readout ----
  readout_kernel<<<B_, 256, 0, stream>>>(h2p, Wlin, blin, out, K2_, H2_);
}
